// SwinTransformerBlock_65180423685489
// MI455X (gfx1250) — compile-verified, matrix operand order checked by reading, not run on hardware
//
#include <hip/hip_runtime.h>
#include <hip/hip_bf16.h>
#include <cstdint>
#include <cstddef>

typedef __bf16 bf16_t;
typedef __bf16 v16bf __attribute__((ext_vector_type(16)));
typedef float  v8f   __attribute__((ext_vector_type(8)));

#define WMMA_BF16(a, b, c) \
    __builtin_amdgcn_wmma_f32_16x16x32_bf16(false, (a), false, (b), (short)0, (c), false, false)

union FragU { v16bf v; uint4 u[2]; };
union Pack8 { uint4 u; bf16_t h[8]; };

// ---------------------------------------------------------------------------
// LDS fragment loader, CDNA5 16-bit A/B striping: lane&15 = M/N row,
// lane>>4 selects K octet (halves 0..7 -> K kh*8.., halves 8..15 -> K 16+kh*8).
// ---------------------------------------------------------------------------
__device__ __forceinline__ v16bf frag_ld(const bf16_t* __restrict__ base,
                                         int row0, int stride, int koff, int lane) {
    int r  = row0 + (lane & 15);
    int kh = (lane >> 4) << 3;
    FragU f;
    f.u[0] = *(const uint4*)(base + r * stride + koff + kh);
    f.u[1] = *(const uint4*)(base + r * stride + koff + 16 + kh);
    return f.v;
}

__device__ __forceinline__ float gelu_exact(float x) {
    return 0.5f * x * (1.0f + erff(x * 0.70710678118654752f));
}

// ---------------------------------------------------------------------------
// Tensor Data Mover: 2D tile load Global->LDS (bf16 elements).
// Builds the D# per CDNA5 ISA:
//   group0: w0=count:1, w1=lds_addr, w2=ga[31:0], w3=ga[56:32]|type=2<<30
//   group1: w0=data_size(1=2B)<<16, w1=tensor_dim0<<16, w2=tensor_dim1<<16,
//           w3=tile_dim0<<16, w4=tile_dim1, w5=tensor_dim0_stride, w6=w7=0
// 2D tensor -> 2-group form (VADDR2/VADDR3 = NULL).  Completion via TENSORcnt.
// ---------------------------------------------------------------------------
__device__ __forceinline__ void tdm_load_2d_bf16(uint32_t lds_addr, uint64_t gaddr,
                                                 uint32_t rows, uint32_t rowlen,
                                                 uint32_t stride_elts) {
    asm volatile(
        "s_mov_b32 s8,  1\n\t"
        "s_mov_b32 s9,  %0\n\t"
        "s_mov_b32 s10, %1\n\t"
        "s_mov_b32 s11, %2\n\t"
        "s_mov_b32 s16, 0x10000\n\t"
        "s_mov_b32 s17, %3\n\t"
        "s_mov_b32 s18, %4\n\t"
        "s_mov_b32 s19, %3\n\t"
        "s_mov_b32 s20, %5\n\t"
        "s_mov_b32 s21, %6\n\t"
        "s_mov_b32 s22, 0\n\t"
        "s_mov_b32 s23, 0\n\t"
        "tensor_load_to_lds s[8:11], s[16:23]\n\t"
        :
        : "s"(lds_addr),
          "s"((uint32_t)gaddr),
          "s"(((uint32_t)(gaddr >> 32) & 0x01FFFFFFu) | 0x80000000u),
          "s"(rowlen << 16),
          "s"(rows << 16),
          "s"(rows),
          "s"(stride_elts)
        : "s8", "s9", "s10", "s11", "s16", "s17", "s18", "s19",
          "s20", "s21", "s22", "s23", "memory");
}

// ---------------------------------------------------------------------------
// Generic WMMA GEMM:  out = epilogue(A[M,K] @ W[N,K]^T + bias)
// Tile BM=128, BN=128, BK=32.  8 waves (4 M-strips x 2 N-strips); each wave
// owns 32x64 -> 2x4 v_wmma_f32_16x16x32_bf16 accumulators, 8 WMMAs / K-slab.
// W is pre-converted bf16 (L2-resident).  A is bf16 or fp32 (converted).
// ---------------------------------------------------------------------------
enum {
    EP_BF16_BIAS = 0,        // bf16 out = acc + bias
    EP_F32_BIAS,             // f32  out = acc + bias
    EP_BF16_BIAS_GELU,       // bf16 out = gelu(acc + bias)
    EP_BF16_BIAS_RES_GELU,   // bf16 out = gelu(acc + bias + res)
    EP_F32_BIAS_RES,         // f32  out = acc + bias + res
    EP_F32_BIAS_GAMMA_RES    // f32  out = res + (acc + bias) * gamma[col]
};

template <typename AT, int EPI>
__global__ __launch_bounds__(256)
void gemm_kernel(const AT* __restrict__ A, const bf16_t* __restrict__ W,
                 const float* __restrict__ bias, const float* __restrict__ res,
                 const float* __restrict__ gamma, void* __restrict__ outp,
                 int M, int N, int K) {
    __shared__ bf16_t lds_a[128 * 32];
    __shared__ bf16_t lds_b[128 * 32];

    const int tid  = threadIdx.x;
    const int lane = tid & 31;
    const int wid  = tid >> 5;
    const int wm   = wid & 3;   // 4 wave-rows of 32
    const int wn   = wid >> 2;  // 2 wave-cols of 64
    const int bm   = blockIdx.x * 128;
    const int bn   = blockIdx.y * 128;

    v8f acc[2][4] = {};

    const int ksteps = K >> 5;
    for (int kt = 0; kt < ksteps; ++kt) {
        // ---- stage A tile (128 x 32) ----
#pragma unroll
        for (int i = 0; i < 2; ++i) {
            int idx = tid + i * 256;
            int row = idx >> 2;
            int cg  = (idx & 3) << 3;
            int gm  = bm + row;
            Pack8 p;
            if (gm < M) {
                const AT* src = A + (size_t)gm * K + (kt << 5) + cg;
                if constexpr (sizeof(AT) == 2) {
                    p.u = *(const uint4*)src;
                } else {
                    float4 f0 = *(const float4*)(src);
                    float4 f1 = *(const float4*)(src + 4);
                    p.h[0] = (bf16_t)f0.x; p.h[1] = (bf16_t)f0.y;
                    p.h[2] = (bf16_t)f0.z; p.h[3] = (bf16_t)f0.w;
                    p.h[4] = (bf16_t)f1.x; p.h[5] = (bf16_t)f1.y;
                    p.h[6] = (bf16_t)f1.z; p.h[7] = (bf16_t)f1.w;
                }
                if (kt + 1 < ksteps)  // -> global_prefetch_b8
                    __builtin_prefetch(A + (size_t)gm * K + ((kt + 1) << 5) + cg, 0, 3);
            } else {
                p.u = make_uint4(0u, 0u, 0u, 0u);
            }
            *(uint4*)(lds_a + row * 32 + cg) = p.u;
        }
        // ---- stage W tile (128 x 32), already bf16 ----
#pragma unroll
        for (int i = 0; i < 2; ++i) {
            int idx = tid + i * 256;
            int row = idx >> 2;
            int cg  = (idx & 3) << 3;
            int gn  = bn + row;
            Pack8 p;
            if (gn < N) p.u = *(const uint4*)(W + (size_t)gn * K + (kt << 5) + cg);
            else        p.u = make_uint4(0u, 0u, 0u, 0u);
            *(uint4*)(lds_b + row * 32 + cg) = p.u;
        }
        __syncthreads();

        v16bf a0 = frag_ld(lds_a, wm * 32 + 0,  32, 0, lane);
        v16bf a1 = frag_ld(lds_a, wm * 32 + 16, 32, 0, lane);
#pragma unroll
        for (int ni = 0; ni < 4; ++ni) {
            v16bf b = frag_ld(lds_b, wn * 64 + ni * 16, 32, 0, lane);
            acc[0][ni] = WMMA_BF16(a0, b, acc[0][ni]);
            acc[1][ni] = WMMA_BF16(a1, b, acc[1][ni]);
        }
        __syncthreads();
    }

    // ---- epilogue: C/D layout -> lane&15 = N, VGPR r -> M = r + 8*(lane>>4)
#pragma unroll
    for (int mi = 0; mi < 2; ++mi)
#pragma unroll
        for (int ni = 0; ni < 4; ++ni) {
            int col = bn + wn * 64 + ni * 16 + (lane & 15);
            int rb  = bm + wm * 32 + mi * 16 + ((lane >> 4) << 3);
            if (col >= N) continue;
            float bv = bias[col];
#pragma unroll
            for (int r = 0; r < 8; ++r) {
                int row = rb + r;
                if (row >= M) continue;
                float v = acc[mi][ni][r] + bv;
                size_t oi = (size_t)row * N + col;
                if constexpr (EPI == EP_BF16_BIAS)
                    ((bf16_t*)outp)[oi] = (bf16_t)v;
                else if constexpr (EPI == EP_F32_BIAS)
                    ((float*)outp)[oi] = v;
                else if constexpr (EPI == EP_BF16_BIAS_GELU)
                    ((bf16_t*)outp)[oi] = (bf16_t)gelu_exact(v);
                else if constexpr (EPI == EP_BF16_BIAS_RES_GELU)
                    ((bf16_t*)outp)[oi] = (bf16_t)gelu_exact(v + res[oi]);
                else if constexpr (EPI == EP_F32_BIAS_RES)
                    ((float*)outp)[oi] = v + res[oi];
                else if constexpr (EPI == EP_F32_BIAS_GAMMA_RES)
                    ((float*)outp)[oi] = res[oi] + v * gamma[col];
            }
        }
}

// ---------------------------------------------------------------------------
// fp32 -> bf16 elementwise (weight pre-conversion, one shot per launch)
// ---------------------------------------------------------------------------
__global__ __launch_bounds__(256)
void cvt_bf16_kernel(const float* __restrict__ src, bf16_t* __restrict__ dst, int n) {
    int t = blockIdx.x * 256 + threadIdx.x;
    if (t < n) dst[t] = (bf16_t)src[t];
}

// ---------------------------------------------------------------------------
// Fused attention bias:  bias[wimg][head][64][64] = rpb[relidx(m,n),head]
//                        + shift_mask[wimg,m,n]  (zero-padded past 49)
// ---------------------------------------------------------------------------
__global__ __launch_bounds__(256)
void bias_prep_kernel(const float* __restrict__ rpb, const float* __restrict__ mask,
                      float* __restrict__ bb) {
    int t    = blockIdx.x * 256 + threadIdx.x;  // 64*8*64*64
    int col  = t & 63;
    int row  = (t >> 6) & 63;
    int head = (t >> 12) & 7;
    int wimg = t >> 15;
    float v = 0.0f;
    if (row < 49 && col < 49) {
        int ri = row / 7, ci = row % 7, rj = col / 7, cj = col % 7;
        int idx = (ri - rj + 6) * 13 + (ci - cj + 6);
        v = rpb[idx * 8 + head] + mask[((size_t)wimg * 49 + row) * 49 + col];
    }
    bb[t] = v;
}

// ---------------------------------------------------------------------------
// LayerNorm over C=256 (one block / token) with wave-shuffle reductions.
// SHIFTWIN=1 fuses cyclic shift + window partition into the input gather.
// ---------------------------------------------------------------------------
template <int SHIFTWIN>
__global__ __launch_bounds__(256)
void ln_kernel(const float* __restrict__ x, const float* __restrict__ g,
               const float* __restrict__ b, bf16_t* __restrict__ out) {
    int row = blockIdx.x;
    int c   = threadIdx.x;
    int inrow;
    if constexpr (SHIFTWIN) {
        int n    = row % 49;
        int wini = (row / 49) & 63;
        int bb   = row / (49 * 64);
        int h    = (wini >> 3) * 7 + n / 7;
        int w    = (wini & 7) * 7 + n % 7;
        int hs   = (h + 3) % 56;   // roll(-3)
        int ws2  = (w + 3) % 56;
        inrow = bb * 3136 + hs * 56 + ws2;
    } else {
        inrow = row;
    }
    float v = x[(size_t)inrow * 256 + c];

    const int lane = c & 31, wid = c >> 5;
    __shared__ float part[8];
    float s = v;
#pragma unroll
    for (int o = 16; o > 0; o >>= 1) s += __shfl_xor(s, o, 32);
    if (lane == 0) part[wid] = s;
    __syncthreads();
    float tot = 0.0f;
#pragma unroll
    for (int i = 0; i < 8; ++i) tot += part[i];
    float mean = tot * (1.0f / 256.0f);
    float d = v - mean;
    __syncthreads();
    float s2 = d * d;
#pragma unroll
    for (int o = 16; o > 0; o >>= 1) s2 += __shfl_xor(s2, o, 32);
    if (lane == 0) part[wid] = s2;
    __syncthreads();
    tot = 0.0f;
#pragma unroll
    for (int i = 0; i < 8; ++i) tot += part[i];
    float var = tot * (1.0f / 256.0f);
    out[(size_t)row * 256 + c] = (bf16_t)(d * rsqrtf(var + 1e-5f) * g[c] + b[c]);
}

// ---------------------------------------------------------------------------
// Window attention: one wave per (window, head).  N=49 padded to 64, HD=32.
// Q/K tiles staged by the Tensor Data Mover; V loaded + transposed by lanes.
// S = Q K^T (16 WMMAs) + fused precomputed bias, f32 softmax, O = P V (16).
// ---------------------------------------------------------------------------
__global__ __launch_bounds__(32)
void attn_kernel(const bf16_t* __restrict__ qkv, const float* __restrict__ biasb,
                 bf16_t* __restrict__ out) {
    __shared__ bf16_t Qs[64 * 32];
    __shared__ bf16_t Ks[64 * 32];
    __shared__ bf16_t Vt[32 * 64];
    __shared__ float  S[64 * 64];
    __shared__ bf16_t Ps[64 * 64];

    const int lane = threadIdx.x;
    const int wh   = blockIdx.x;
    const int win  = wh >> 3;
    const int head = wh & 7;
    const int wimg = win & 63;

    const bf16_t* base = qkv + (size_t)win * 49 * 768 + head * 32;

    // TDM: rows 0..48 of Q and K (32 bf16 each, row stride 768 elements)
    tdm_load_2d_bf16((uint32_t)(uintptr_t)(void*)Qs, (uint64_t)(uintptr_t)base,
                     49u, 32u, 768u);
    tdm_load_2d_bf16((uint32_t)(uintptr_t)(void*)Ks, (uint64_t)(uintptr_t)(base + 256),
                     49u, 32u, 768u);

    // lanes: zero-pad rows 49..63 of Q/K, load V rows and transpose into Vt
    uint4 z = make_uint4(0u, 0u, 0u, 0u);
    for (int r = lane; r < 64; r += 32) {
        if (r >= 49) {
            uint4* dq = (uint4*)(Qs + r * 32); dq[0] = z; dq[1] = z; dq[2] = z; dq[3] = z;
            uint4* dk = (uint4*)(Ks + r * 32); dk[0] = z; dk[1] = z; dk[2] = z; dk[3] = z;
        }
        union { uint4 u[4]; bf16_t h[32]; } vv;
        vv.u[0] = z; vv.u[1] = z; vv.u[2] = z; vv.u[3] = z;
        if (r < 49) {
            const uint4* pv = (const uint4*)(base + (size_t)r * 768 + 512);
            vv.u[0] = pv[0]; vv.u[1] = pv[1]; vv.u[2] = pv[2]; vv.u[3] = pv[3];
        }
#pragma unroll
        for (int c = 0; c < 32; ++c) Vt[c * 64 + r] = vv.h[c];
    }
    __builtin_amdgcn_s_wait_tensorcnt(0);
    __syncthreads();

    const float scale = 0.17677669529663689f;  // 32^-0.5
    const float* bb = biasb + ((size_t)(wimg * 8 + head) << 12);
#pragma unroll
    for (int mi = 0; mi < 4; ++mi)
#pragma unroll
        for (int ni = 0; ni < 4; ++ni) {
            v8f acc = {};
            v16bf a = frag_ld(Qs, mi * 16, 32, 0, lane);
            v16bf b = frag_ld(Ks, ni * 16, 32, 0, lane);
            acc = WMMA_BF16(a, b, acc);
            int col = ni * 16 + (lane & 15);
            int rb  = mi * 16 + ((lane >> 4) << 3);
#pragma unroll
            for (int r = 0; r < 8; ++r) {
                int row = rb + r;
                S[row * 64 + col] = acc[r] * scale + bb[row * 64 + col];
            }
        }
    __syncthreads();

    for (int r = lane; r < 64; r += 32) {
        if (r < 49) {
            float mx = -1e30f;
            for (int n = 0; n < 49; ++n) mx = fmaxf(mx, S[r * 64 + n]);
            float sum = 0.0f;
            for (int n = 0; n < 49; ++n) {
                float e = __expf(S[r * 64 + n] - mx);
                S[r * 64 + n] = e;
                sum += e;
            }
            float inv = 1.0f / sum;
            for (int n = 0; n < 49; ++n) Ps[r * 64 + n] = (bf16_t)(S[r * 64 + n] * inv);
            for (int n = 49; n < 64; ++n) Ps[r * 64 + n] = (bf16_t)0.0f;
        } else {
            for (int n = 0; n < 64; ++n) Ps[r * 64 + n] = (bf16_t)0.0f;
        }
    }
    __syncthreads();

    bf16_t* obase = out + (size_t)win * 49 * 256 + head * 32;
#pragma unroll
    for (int mi = 0; mi < 4; ++mi) {
        v8f o0 = {}, o1 = {};
#pragma unroll
        for (int ks = 0; ks < 2; ++ks) {
            v16bf p  = frag_ld(Ps, mi * 16, 64, ks * 32, lane);
            v16bf w0 = frag_ld(Vt, 0,  64, ks * 32, lane);
            v16bf w1 = frag_ld(Vt, 16, 64, ks * 32, lane);
            o0 = WMMA_BF16(p, w0, o0);
            o1 = WMMA_BF16(p, w1, o1);
        }
        int rb = mi * 16 + ((lane >> 4) << 3);
        int c0 = lane & 15;
#pragma unroll
        for (int r = 0; r < 8; ++r) {
            int row = rb + r;
            if (row < 49) {
                obase[(size_t)row * 256 + c0]      = (bf16_t)o0[r];
                obase[(size_t)row * 256 + 16 + c0] = (bf16_t)o1[r];
            }
        }
    }
}

// ---------------------------------------------------------------------------
// Window-reverse + un-shift + residual add:  x1 = x + reverse(projtmp)
// ---------------------------------------------------------------------------
__global__ __launch_bounds__(256)
void unwindow_res_kernel(const float* __restrict__ x, const float* __restrict__ tmp,
                         float* __restrict__ out) {
    size_t t = (size_t)blockIdx.x * 256 + threadIdx.x;
    int c    = (int)(t & 255);
    size_t l = (t >> 8) % 3136;
    size_t b = t / ((size_t)256 * 3136);
    int h = (int)(l / 56), w = (int)(l % 56);
    int hs  = (h + 53) % 56;  // roll(+3)
    int ws2 = (w + 53) % 56;
    int win = (hs / 7) * 8 + (ws2 / 7);
    int n   = (hs % 7) * 7 + (ws2 % 7);
    out[t] = x[t] + tmp[(((size_t)b * 64 + win) * 49 + n) * 256 + c];
}

// ---------------------------------------------------------------------------
// Mona multi-scale depthwise convs:  mix = (dw3 + dw5 + dw7)/3 + h
// ---------------------------------------------------------------------------
__global__ __launch_bounds__(256)
void mona_conv_kernel(const float* __restrict__ hsrc,
                      const float* __restrict__ w3, const float* __restrict__ b3,
                      const float* __restrict__ w5, const float* __restrict__ b5,
                      const float* __restrict__ w7, const float* __restrict__ b7,
                      float* __restrict__ mix) {
    size_t t  = (size_t)blockIdx.x * 256 + threadIdx.x;
    int ch    = (int)(t & 63);
    size_t px = t >> 6;
    int w = (int)(px % 56);
    int h = (int)((px / 56) % 56);
    size_t b = px / 3136;
    const float* img = hsrc + b * 3136 * 64 + ch;

    float a3 = 0.0f, a5 = 0.0f, a7 = 0.0f;
    for (int ky = 0; ky < 7; ++ky) {
        int yy = h + ky - 3;
        if (yy < 0 || yy >= 56) continue;
        for (int kx = 0; kx < 7; ++kx) {
            int xx = w + kx - 3;
            if (xx < 0 || xx >= 56) continue;
            float pv = img[((size_t)yy * 56 + xx) * 64];
            a7 += pv * w7[ch * 49 + ky * 7 + kx];
            if (ky >= 1 && ky <= 5 && kx >= 1 && kx <= 5)
                a5 += pv * w5[ch * 25 + (ky - 1) * 5 + (kx - 1)];
            if (ky >= 2 && ky <= 4 && kx >= 2 && kx <= 4)
                a3 += pv * w3[ch * 9 + (ky - 2) * 3 + (kx - 2)];
        }
    }
    float center = img[((size_t)h * 56 + w) * 64];
    mix[t] = (a3 + b3[ch] + a5 + b5[ch] + a7 + b7[ch]) * (1.0f / 3.0f) + center;
}

// ---------------------------------------------------------------------------
// Host side
// ---------------------------------------------------------------------------
static const int MROWS = 32 * 3136;  // 100352 tokens (== B*NW*49)

static inline dim3 gemm_grid(int M, int N) { return dim3((M + 127) / 128, (N + 127) / 128); }

static inline bf16_t* cvt_w(const float* src, int n, char*& cur, hipStream_t stream) {
    bf16_t* dst = (bf16_t*)cur;
    cur += ((size_t)n * 2 + 255) & ~(size_t)255;
    cvt_bf16_kernel<<<(n + 255) / 256, 256, 0, stream>>>(src, dst, n);
    return dst;
}

static void run_mona(const float* xin, float* xout, void* const* mp,
                     const bf16_t* p1w, const bf16_t* pcw, const bf16_t* p2w,
                     bf16_t* xb, float* h64, float* mixb, bf16_t* g64,
                     hipStream_t stream) {
    const float* ng  = (const float*)mp[0];
    const float* nb  = (const float*)mp[1];
    const float* p1b = (const float*)mp[3];
    const float* c1w = (const float*)mp[4];
    const float* c1b = (const float*)mp[5];
    const float* c2w = (const float*)mp[6];
    const float* c2b = (const float*)mp[7];
    const float* c3w = (const float*)mp[8];
    const float* c3b = (const float*)mp[9];
    const float* pcb = (const float*)mp[11];
    const float* p2b = (const float*)mp[13];
    const float* gma = (const float*)mp[14];

    ln_kernel<0><<<MROWS, 256, 0, stream>>>(xin, ng, nb, xb);
    gemm_kernel<bf16_t, EP_F32_BIAS><<<gemm_grid(MROWS, 64), 256, 0, stream>>>(
        xb, p1w, p1b, nullptr, nullptr, h64, MROWS, 64, 256);
    mona_conv_kernel<<<(MROWS * 64) / 256, 256, 0, stream>>>(
        h64, c1w, c1b, c2w, c2b, c3w, c3b, mixb);
    gemm_kernel<float, EP_BF16_BIAS_RES_GELU><<<gemm_grid(MROWS, 64), 256, 0, stream>>>(
        mixb, pcw, pcb, mixb, nullptr, g64, MROWS, 64, 64);
    gemm_kernel<bf16_t, EP_F32_BIAS_GAMMA_RES><<<gemm_grid(MROWS, 256), 256, 0, stream>>>(
        g64, p2w, p2b, xin, gma, xout, MROWS, 256, 64);
}

extern "C" void kernel_launch(void* const* d_in, const int* in_sizes, int n_in,
                              void* d_out, int out_size, void* d_ws, size_t ws_size,
                              hipStream_t stream) {
    (void)in_sizes; (void)n_in; (void)out_size; (void)ws_size;

    const float* x      = (const float*)d_in[0];
    const float* mask   = (const float*)d_in[1];
    const float* n1g    = (const float*)d_in[2];
    const float* n1b    = (const float*)d_in[3];
    const float* qkv_w  = (const float*)d_in[4];
    const float* qkv_b  = (const float*)d_in[5];
    const float* rpb    = (const float*)d_in[6];
    const float* proj_w = (const float*)d_in[7];
    const float* proj_b = (const float*)d_in[8];
    const float* n2g    = (const float*)d_in[9];
    const float* n2b    = (const float*)d_in[10];
    const float* fc1_w  = (const float*)d_in[11];
    const float* fc1_b  = (const float*)d_in[12];
    const float* fc2_w  = (const float*)d_in[13];
    const float* fc2_b  = (const float*)d_in[14];
    void* const* m1 = d_in + 15;
    void* const* m2 = d_in + 30;

    const size_t M = (size_t)MROWS;
    char* cur = (char*)d_ws;
    auto carve = [&cur](size_t bytes) -> char* {
        char* p = cur;
        cur += (bytes + 255) & ~(size_t)255;
        return p;
    };
    bf16_t* big   = (bf16_t*)carve(M * 1024 * 2);  // qkv (768) -> fc1 hidden (1024)
    bf16_t* bufA  = (bf16_t*)carve(M * 256 * 2);   // xw -> attn_out -> LN scratch
    float*  bufF1 = (float*) carve(M * 256 * 4);   // proj tmp -> x2 (post-mona1)
    float*  bufF2 = (float*) carve(M * 256 * 4);   // x1 (post-attn residual)
    float*  h64   = (float*) carve(M * 64 * 4);
    float*  mixb  = (float*) carve(M * 64 * 4);
    bf16_t* g64   = (bf16_t*)carve(M * 64 * 2);
    float*  biasb = (float*) carve((size_t)64 * 8 * 64 * 64 * 4);  // fused attn bias

    // one-shot weight conversions (f32 -> bf16) and fused-bias precompute
    bf16_t* wqkv  = cvt_w(qkv_w, 768 * 256, cur, stream);
    bf16_t* wproj = cvt_w(proj_w, 256 * 256, cur, stream);
    bf16_t* wfc1  = cvt_w(fc1_w, 1024 * 256, cur, stream);
    bf16_t* wfc2  = cvt_w(fc2_w, 256 * 1024, cur, stream);
    bf16_t* wp1a  = cvt_w((const float*)m1[2], 64 * 256, cur, stream);
    bf16_t* wpca  = cvt_w((const float*)m1[10], 64 * 64, cur, stream);
    bf16_t* wp2a  = cvt_w((const float*)m1[12], 256 * 64, cur, stream);
    bf16_t* wp1b  = cvt_w((const float*)m2[2], 64 * 256, cur, stream);
    bf16_t* wpcb  = cvt_w((const float*)m2[10], 64 * 64, cur, stream);
    bf16_t* wp2b  = cvt_w((const float*)m2[12], 256 * 64, cur, stream);
    bias_prep_kernel<<<(64 * 8 * 64 * 64) / 256, 256, 0, stream>>>(rpb, mask, biasb);

    // 1) LN1 + shift + window partition -> bf16
    ln_kernel<1><<<MROWS, 256, 0, stream>>>(x, n1g, n1b, bufA);
    // 2) QKV GEMM (bf16 WMMA)
    gemm_kernel<bf16_t, EP_BF16_BIAS><<<gemm_grid(MROWS, 768), 256, 0, stream>>>(
        bufA, wqkv, qkv_b, nullptr, nullptr, big, MROWS, 768, 256);
    // 3) window attention (TDM-staged Q/K, 32 WMMAs per window-head)
    attn_kernel<<<32 * 64 * 8, 32, 0, stream>>>(big, biasb, bufA);
    // 4) output projection
    gemm_kernel<bf16_t, EP_F32_BIAS><<<gemm_grid(MROWS, 256), 256, 0, stream>>>(
        bufA, wproj, proj_b, nullptr, nullptr, bufF1, MROWS, 256, 256);
    // 5) window reverse + un-shift + residual
    unwindow_res_kernel<<<MROWS, 256, 0, stream>>>(x, bufF1, bufF2);
    // 6) Mona adapter 1:  bufF2 (x1) -> bufF1 (x2)
    run_mona(bufF2, bufF1, m1, wp1a, wpca, wp2a, bufA, h64, mixb, g64, stream);
    // 7) MLP: LN2 -> FC1+gelu -> FC2 + residual, write x3 into d_out
    ln_kernel<0><<<MROWS, 256, 0, stream>>>(bufF1, n2g, n2b, bufA);
    gemm_kernel<bf16_t, EP_BF16_BIAS_GELU><<<gemm_grid(MROWS, 1024), 256, 0, stream>>>(
        bufA, wfc1, fc1_b, nullptr, nullptr, big, MROWS, 1024, 256);
    gemm_kernel<bf16_t, EP_F32_BIAS_RES><<<gemm_grid(MROWS, 256), 256, 0, stream>>>(
        big, wfc2, fc2_b, bufF1, nullptr, (float*)d_out, MROWS, 256, 1024);
    // 8) Mona adapter 2: in-place on d_out
    run_mona((const float*)d_out, (float*)d_out, m2, wp1b, wpcb, wp2b,
             bufA, h64, mixb, g64, stream);
}
